// NodeDegreeExpansion_7739531067656
// MI455X (gfx1250) — matrix operand order repaired and checked by reading, not compiled, not run on hardware
//
#include <hip/hip_runtime.h>

typedef __bf16 bf16;
typedef __attribute__((ext_vector_type(16))) __bf16 v16bf;
typedef __attribute__((ext_vector_type(8)))  float  v8f;

#define MULI 32
#define NSPECI 10
#define NGRAPHI 16

// ---------------- WMMA fragment layout helpers (CDNA5 ISA 7.12.2) ----------------

// A-matrix 16x32 bf16: lane L holds row (L&15); element slot e -> K index:
//   lanes 0-15 : e0..7 -> K 0..7,  e8..15 -> K 16..23
//   lanes 16-31: e0..7 -> K 8..15, e8..15 -> K 24..31
__device__ __forceinline__ int a_k_idx(int L, int e) {
  int hi = (L >> 4) & 1;
  return (e < 8) ? (e + (hi ? 8 : 0)) : ((e - 8) + (hi ? 24 : 16));
}

// fragment-major storage index for a 32(K) x 32(N) weight panel chunk:
// element (k, col w) placed so lane L / tile t reads 16 contiguous bf16 (one v16bf)
__device__ __forceinline__ int bfrag_index(int k, int w) {
  int kr = k & 31;
  int chunk = k >> 5;
  int t = w >> 4;
  int col = w & 15;
  int L = col + (((kr >> 3) & 1) << 4);
  int e = (kr & 7) + ((kr >> 4) << 3);
  return (chunk * 2 + t) * 512 + L * 16 + e;
}

__device__ __forceinline__ v16bf load_bfrag(const bf16* __restrict__ W, int L, int t, int chunk) {
  return ((const v16bf*)W)[(chunk * 2 + t) * 32 + L];
}

__device__ __forceinline__ v8f wmma_bf16(v16bf a, v16bf b, v8f c) {
  return __builtin_amdgcn_wmma_f32_16x16x32_bf16(false, a, false, b, (short)0, c, false, false);
}

// Build A fragment (16 nodes x 32 ch) from LDS-staged features, scalar part
__device__ __forceinline__ v16bf afrag_s(const float* f, int L) {
  v16bf a; int r = L & 15;
  #pragma unroll
  for (int e = 0; e < 16; ++e) a[e] = (bf16)f[r * 128 + a_k_idx(L, e)];
  return a;
}
// vector part, component c (memory layout 32 + u*3 + c)
__device__ __forceinline__ v16bf afrag_v(const float* f, int L, int c) {
  v16bf a; int r = L & 15;
  #pragma unroll
  for (int e = 0; e < 16; ++e) a[e] = (bf16)f[r * 128 + 32 + a_k_idx(L, e) * 3 + c];
  return a;
}

// ---------------- Kernel 1: fold + convert weights, zero stats ----------------
__global__ __launch_bounds__(256)
void e3_prep(const float* __restrict__ w000, const float* __restrict__ w110,
             const float* __restrict__ w011, const float* __restrict__ w101,
             const float* __restrict__ lp_w0, const float* __restrict__ lp_w1,
             const float* __restrict__ hl_w0, const float* __restrict__ hl_w1,
             const float* __restrict__ ul_w0, const float* __restrict__ ul_w1,
             const float* __restrict__ sk_w0, const float* __restrict__ sk_w1,
             bf16* __restrict__ Wz0, bf16* __restrict__ Wz1,
             bf16* __restrict__ Wlin, float* __restrict__ stats)
{
  const float invlp = 0.17677669529663689f;                       // 1/sqrt(32)
  const float tp_sc = (1.0f / 32.0f) * 0.7071067811865476f * invlp;
  int idx = blockIdx.x * 256 + threadIdx.x;
  if (idx < 65536) {
    int k = idx >> 5, w = idx & 31;
    const float* wp = w000; int kk = k; float sc = tp_sc;
    if (k >= 1024) { wp = w110; kk = k - 1024; sc *= 0.57735026918962576f; }
    const float* row = wp + (size_t)kk * 32;                      // (u*32+v)*32
    float acc = 0.f;
    #pragma unroll 8
    for (int t = 0; t < 32; ++t) acc += row[t] * lp_w0[t * 32 + w];
    Wz0[bfrag_index(k, w)] = (bf16)(acc * sc);
  } else if (idx < 131072) {
    int j = idx - 65536;
    int k = j >> 5, w = j & 31;
    const float* wp = w011; int kk = k;
    if (k >= 1024) { wp = w101; kk = k - 1024; }
    const float* row = wp + (size_t)kk * 32;
    float acc = 0.f;
    #pragma unroll 8
    for (int t = 0; t < 32; ++t) acc += row[t] * lp_w1[t * 32 + w];
    Wz1[bfrag_index(k, w)] = (bf16)(acc * tp_sc);
  } else if (idx < 137216) {
    int j = idx - 131072;
    int m = j >> 10, e = j & 1023;
    const float* src = (m == 0) ? hl_w0 : (m == 1) ? hl_w1 : (m == 2) ? ul_w0
                     : (m == 3) ? ul_w1 : (m == 4) ? sk_w0 : sk_w1;
    Wlin[m * 1024 + bfrag_index(e >> 5, e & 31)] = (bf16)(src[e] * invlp);
  } else if (idx < 137216 + NGRAPHI * 4) {
    stats[idx - 137216] = 0.f;
  }
}

// ---------------- Kernel 2: per 16-node tile, one wave ----------------
__global__ __launch_bounds__(32)
void e3_main_tile(const float* __restrict__ hid,    // [N,128] hidden slice
                  const float* __restrict__ up,     // [N,128]
                  const float* __restrict__ sc_w0,  // [32,10,32]
                  const float* __restrict__ sc_w1,  // [32,10,32]
                  const float* __restrict__ lp_b0,  // [32]
                  const float* __restrict__ sk_b0,  // [32]
                  const int* __restrict__ species,
                  const int* __restrict__ batch,
                  const bf16* __restrict__ Wz0, const bf16* __restrict__ Wz1,
                  const bf16* __restrict__ Wlin,
                  float* __restrict__ z_pre, float* __restrict__ sk_out,
                  float* __restrict__ stats, int N)
{
  __shared__ __align__(16) float featH[16 * 128];    // raw hidden (alive to the end)
  __shared__ __align__(16) float featU[16 * 128];    // raw up; reused as zbuf after TP
  // interleaved post-linear features: {scalar, vec0, vec1, vec2} per (row, ch)
  __shared__ __align__(16) float postH[16 * 128];    // hidden' : H[r][u] float4
  __shared__ __align__(16) float postU[16 * 128];    // up'     : U[r][v] float4

  float (*zbuf)[128] = (float (*)[128])featU;        // alias (dead after linears)

  const int L  = threadIdx.x;
  const int hi = (L >> 4) & 1;
  const int lr = L & 15;
  const int base = blockIdx.x * 16;
  const bool tile_full = (base + 16 <= N);           // uniform across wave

  // ---- stage features into LDS (float4)
  {
    const float4* s0 = (const float4*)hid + (size_t)base * 32;
    const float4* s1 = (const float4*)up  + (size_t)base * 32;
    float4* d0 = (float4*)featH;
    float4* d1 = (float4*)featU;
    if (tile_full) {
      for (int j = L; j < 512; j += 32) { d0[j] = s0[j]; d1[j] = s1[j]; }
    } else {
      for (int j = L; j < 512; j += 32) {
        int node = base + (j >> 5);
        float4 z; z.x = z.y = z.z = z.w = 0.f;
        d0[j] = (node < N) ? s0[j] : z;
        d1[j] = (node < N) ? s1[j] : z;
      }
    }
  }
  __syncthreads();

  // ---- small linears as K=32 WMMAs ----
  {
    v16bf a_s = afrag_s(featH, L);
    #pragma unroll
    for (int t = 0; t < 2; ++t) {                 // shp -> H comp 0
      v8f d = {};
      d = wmma_bf16(a_s, load_bfrag(Wlin + 0 * 1024, L, t, 0), d);
      #pragma unroll
      for (int i = 0; i < 8; ++i) postH[(i + 8 * hi) * 128 + (lr + 16 * t) * 4 + 0] = d[i];
    }
    #pragma unroll
    for (int t = 0; t < 2; ++t) {                 // sk scalars (+bias) -> global
      v8f d = {};
      d = wmma_bf16(a_s, load_bfrag(Wlin + 4 * 1024, L, t, 0), d);
      int w = lr + 16 * t;
      float bias = sk_b0[w];
      if (tile_full) {
        #pragma unroll
        for (int i = 0; i < 8; ++i)
          sk_out[(size_t)(base + i + 8 * hi) * 128 + w] = d[i] + bias;
      } else {
        #pragma unroll
        for (int i = 0; i < 8; ++i) {
          int node = base + i + 8 * hi;
          if (node < N) sk_out[(size_t)node * 128 + w] = d[i] + bias;
        }
      }
    }
    #pragma unroll
    for (int c = 0; c < 3; ++c) {
      v16bf a_v = afrag_v(featH, L, c);
      #pragma unroll
      for (int t = 0; t < 2; ++t) {               // vhp -> H comp 1+c
        v8f d = {};
        d = wmma_bf16(a_v, load_bfrag(Wlin + 1 * 1024, L, t, 0), d);
        #pragma unroll
        for (int i = 0; i < 8; ++i)
          postH[(i + 8 * hi) * 128 + (lr + 16 * t) * 4 + 1 + c] = d[i];
      }
      #pragma unroll
      for (int t = 0; t < 2; ++t) {               // sk vectors -> global
        v8f d = {};
        d = wmma_bf16(a_v, load_bfrag(Wlin + 5 * 1024, L, t, 0), d);
        int w = lr + 16 * t;
        if (tile_full) {
          #pragma unroll
          for (int i = 0; i < 8; ++i)
            sk_out[(size_t)(base + i + 8 * hi) * 128 + 32 + w * 3 + c] = d[i];
        } else {
          #pragma unroll
          for (int i = 0; i < 8; ++i) {
            int node = base + i + 8 * hi;
            if (node < N) sk_out[(size_t)node * 128 + 32 + w * 3 + c] = d[i];
          }
        }
      }
    }
    v16bf u_s = afrag_s(featU, L);
    #pragma unroll
    for (int t = 0; t < 2; ++t) {                 // sup -> U comp 0
      v8f d = {};
      d = wmma_bf16(u_s, load_bfrag(Wlin + 2 * 1024, L, t, 0), d);
      #pragma unroll
      for (int i = 0; i < 8; ++i) postU[(i + 8 * hi) * 128 + (lr + 16 * t) * 4 + 0] = d[i];
    }
    #pragma unroll
    for (int c = 0; c < 3; ++c) {                 // vup -> U comp 1+c
      v16bf u_v = afrag_v(featU, L, c);
      #pragma unroll
      for (int t = 0; t < 2; ++t) {
        v8f d = {};
        d = wmma_bf16(u_v, load_bfrag(Wlin + 3 * 1024, L, t, 0), d);
        #pragma unroll
        for (int i = 0; i < 8; ++i)
          postU[(i + 8 * hi) * 128 + (lr + 16 * t) * 4 + 1 + c] = d[i];
      }
    }
  }
  __syncthreads();    // featU (raw up) is dead from here on -> reuse as zbuf

  // ---- tensor product as outer-product GEMM, K=2048, folded lp linear ----
  // Within each 32-wide K chunk, u = k>>5 is constant -> load H once per chunk,
  // and each slot needs only one float4 from U.
  v8f z0t[2] = {};
  v8f z1t[3][2] = {};
  const int r = lr;
  for (int kc = 0; kc < 64; ++kc) {
    const bool first = (kc < 32);                  // K<1024 : scalar-x path
    const int u = first ? kc : (kc - 32);
    const float4 H = *(const float4*)&postH[r * 128 + u * 4];
    v16bf aP, aQ0, aQ1, aQ2;
    #pragma unroll
    for (int e = 0; e < 16; ++e) {
      int v = a_k_idx(L, e);
      const float4 U = *(const float4*)&postU[r * 128 + v * 4];
      float p, q0, q1, q2;
      if (first) {
        p  = H.x * U.x;
        q0 = H.x * U.y;
        q1 = H.x * U.z;
        q2 = H.x * U.w;
      } else {
        p  = H.y * U.y + H.z * U.z + H.w * U.w;
        q0 = H.y * U.x;
        q1 = H.z * U.x;
        q2 = H.w * U.x;
      }
      aP[e]  = (bf16)p;
      aQ0[e] = (bf16)q0;
      aQ1[e] = (bf16)q1;
      aQ2[e] = (bf16)q2;
    }
    v16bf b00 = load_bfrag(Wz0, L, 0, kc);
    v16bf b01 = load_bfrag(Wz0, L, 1, kc);
    v16bf b10 = load_bfrag(Wz1, L, 0, kc);
    v16bf b11 = load_bfrag(Wz1, L, 1, kc);
    z0t[0]    = wmma_bf16(aP,  b00, z0t[0]);
    z0t[1]    = wmma_bf16(aP,  b01, z0t[1]);
    z1t[0][0] = wmma_bf16(aQ0, b10, z1t[0][0]);
    z1t[0][1] = wmma_bf16(aQ0, b11, z1t[0][1]);
    z1t[1][0] = wmma_bf16(aQ1, b10, z1t[1][0]);
    z1t[1][1] = wmma_bf16(aQ1, b11, z1t[1][1]);
    z1t[2][0] = wmma_bf16(aQ2, b10, z1t[2][0]);
    z1t[2][1] = wmma_bf16(aQ2, b11, z1t[2][1]);
  }

  // ---- spill z tiles to LDS (zbuf aliases featU)
  #pragma unroll
  for (int t = 0; t < 2; ++t)
    #pragma unroll
    for (int i = 0; i < 8; ++i)
      zbuf[i + 8 * hi][lr + 16 * t] = z0t[t][i];
  #pragma unroll
  for (int c = 0; c < 3; ++c)
    #pragma unroll
    for (int t = 0; t < 2; ++t)
      #pragma unroll
      for (int i = 0; i < 8; ++i)
        zbuf[i + 8 * hi][32 + (lr + 16 * t) * 3 + c] = z1t[c][t][i];
  __syncthreads();

  // ---- self-connection (species-gathered weights) + lp bias, write z_pre ----
  // m is wave-uniform: species load is scalar, feature reads are broadcasts,
  // lanes sweep consecutive output channels (coalesced weight gathers).
  const float inv_sc = 0.05590169943749474f;       // 1/sqrt(32*10)
  const float lpb = lp_b0[L];
  for (int m = 0; m < 16; ++m) {
    int node = base + m;
    if (!tile_full && node >= N) continue;
    int sp = species[node];
    // scalar channel w = L
    {
      float acc = zbuf[m][L] + lpb;
      float t = 0.f;
      #pragma unroll 8
      for (int u = 0; u < 32; ++u)
        t += featH[m * 128 + u] * sc_w0[(u * NSPECI + sp) * 32 + L];
      acc += t * inv_sc;
      zbuf[m][L] = acc;
      z_pre[(size_t)node * 128 + L] = acc;
    }
    // vector channels (w = L, c = 0..2) -> 32 + L*3 + c
    #pragma unroll
    for (int c = 0; c < 3; ++c) {
      float acc = zbuf[m][32 + L * 3 + c];
      float t = 0.f;
      #pragma unroll 8
      for (int u = 0; u < 32; ++u)
        t += featH[m * 128 + 32 + u * 3 + c] * sc_w1[(u * NSPECI + sp) * 32 + L];
      acc += t * inv_sc;
      zbuf[m][32 + L * 3 + c] = acc;
      z_pre[(size_t)node * 128 + 32 + L * 3 + c] = acc;
    }
  }
  __syncthreads();

  // ---- per-graph moment accumulation (one lane per node) ----
  if (L < 16) {
    int node = base + L;
    if (node < N) {
      int g = batch[node];
      float ssum = 0.f, ss = 0.f, vv = 0.f;
      #pragma unroll 8
      for (int w = 0; w < 32; ++w) { float x = zbuf[L][w]; ssum += x; ss += x * x; }
      #pragma unroll 8
      for (int j = 32; j < 128; ++j) { float x = zbuf[L][j]; vv += x * x; }
      atomicAdd(&stats[g * 4 + 0], 1.0f);
      atomicAdd(&stats[g * 4 + 1], ssum);
      atomicAdd(&stats[g * 4 + 2], ss);
      atomicAdd(&stats[g * 4 + 3], vv);
    }
  }
}

// ---------------- Kernel 3: LayerNorm apply + skip add ----------------
__global__ __launch_bounds__(256)
void e3_apply(const float* __restrict__ z_pre, const float* __restrict__ sk,
              const float* __restrict__ stats,
              const float* __restrict__ ln_w0, const float* __restrict__ ln_b0,
              const float* __restrict__ ln_w1,
              const int* __restrict__ batch,
              float* __restrict__ out, int N)
{
  int idx = blockIdx.x * 256 + threadIdx.x;
  if (idx >= N * 128) return;
  int node = idx >> 7, ch = idx & 127;
  int g = batch[node];
  float cnt  = fmaxf(stats[g * 4 + 0], 1.f);
  float dv   = 1.f / (cnt * 32.f);
  float mean = stats[g * 4 + 1] * dv;
  float n2s  = fmaxf(stats[g * 4 + 2] * dv - mean * mean, 0.f);
  float n2v  = stats[g * 4 + 3] * dv;
  float z = z_pre[idx];
  float o;
  if (ch < 32) {
    o = (z - mean) * rsqrtf(n2s + 1e-5f) * ln_w0[ch] + ln_b0[ch];
  } else {
    int w = (ch - 32) / 3;
    o = z * rsqrtf(n2v + 1e-5f) * ln_w1[w];
  }
  out[idx] = o + sk[idx];
}

// ---------------- host launcher ----------------
extern "C" void kernel_launch(void* const* d_in, const int* in_sizes, int n_in,
                              void* d_out, int out_size, void* d_ws, size_t ws_size,
                              hipStream_t stream) {
  (void)n_in; (void)out_size; (void)ws_size;
  const float* hid   = (const float*)d_in[0];
  const float* up0   = (const float*)d_in[1];
  const float* hl_w0 = (const float*)d_in[2];
  const float* hl_w1 = (const float*)d_in[3];
  const float* ul_w0 = (const float*)d_in[4];
  const float* ul_w1 = (const float*)d_in[5];
  const float* w000  = (const float*)d_in[6];
  const float* w110  = (const float*)d_in[7];
  const float* w011  = (const float*)d_in[8];
  const float* w101  = (const float*)d_in[9];
  const float* lp_w0 = (const float*)d_in[10];
  const float* lp_b0 = (const float*)d_in[11];
  const float* lp_w1 = (const float*)d_in[12];
  const float* sc_w0 = (const float*)d_in[13];
  const float* sc_w1 = (const float*)d_in[14];
  const float* ln_w0 = (const float*)d_in[15];
  const float* ln_b0 = (const float*)d_in[16];
  const float* ln_w1 = (const float*)d_in[17];
  const float* sk_w0 = (const float*)d_in[18];
  const float* sk_b0 = (const float*)d_in[19];
  const float* sk_w1 = (const float*)d_in[20];
  const int* species = (const int*)d_in[21];
  const int* batch   = (const int*)d_in[22];

  const int N = in_sizes[1] / 128;
  const int B = in_sizes[0] / (N * 128);

  char* ws = (char*)d_ws;
  bf16*  Wz0    = (bf16*)(ws + 0);                       // 2048*32*2B = 128KB
  bf16*  Wz1    = (bf16*)(ws + (size_t)131072);          // 128KB
  bf16*  Wlin   = (bf16*)(ws + (size_t)262144);          // 6*1024*2B = 12KB
  float* stats  = (float*)(ws + (size_t)274432);         // 16*4 floats
  float* z_pre  = (float*)(ws + (size_t)524288);         // N*128 f32
  float* sk_buf = (float*)(ws + (size_t)524288 + (size_t)N * 512);

  const int prep_tot = 137216 + NGRAPHI * 4;
  dim3 pgrid((prep_tot + 255) / 256);
  dim3 tgrid((N + 15) / 16);
  dim3 agrid((N * 128 + 255) / 256);

  for (int b = 0; b < B; ++b) {
    const float* fh = hid + (size_t)b * N * 128;
    const float* up = (b == 0) ? up0 : ((const float*)d_out + (size_t)(b - 1) * N * 128);
    float* outb = (float*)d_out + (size_t)b * N * 128;

    e3_prep<<<pgrid, 256, 0, stream>>>(
        w000 + (size_t)b * 32768, w110 + (size_t)b * 32768,
        w011 + (size_t)b * 32768, w101 + (size_t)b * 32768,
        lp_w0 + (size_t)b * 1024, lp_w1 + (size_t)b * 1024,
        hl_w0 + (size_t)b * 1024, hl_w1 + (size_t)b * 1024,
        ul_w0 + (size_t)b * 1024, ul_w1 + (size_t)b * 1024,
        sk_w0 + (size_t)b * 1024, sk_w1 + (size_t)b * 1024,
        Wz0, Wz1, Wlin, stats);

    e3_main_tile<<<tgrid, 32, 0, stream>>>(
        fh, up,
        sc_w0 + (size_t)b * MULI * NSPECI * MULI,
        sc_w1 + (size_t)b * MULI * NSPECI * MULI,
        lp_b0 + (size_t)b * 32, sk_b0 + (size_t)b * 32,
        species, batch, Wz0, Wz1, Wlin,
        z_pre, sk_buf, stats, N);

    e3_apply<<<agrid, 256, 0, stream>>>(
        z_pre, sk_buf, stats,
        ln_w0 + (size_t)b * 32, ln_b0 + (size_t)b * 32, ln_w1 + (size_t)b * 32,
        batch, outb, N);
  }
}
